// MultiAttention_18691697672248
// MI455X (gfx1250) — compile-verified, matrix operand order checked by reading, not run on hardware
//
#include <hip/hip_runtime.h>

// ---------------------------------------------------------------------------
// MHA block for MI455X (gfx1250, wave32, WMMA).
//   B=2, S=2048, D=1024, H=16, DH=64
// Pipeline: f32->f16 cvt -> 3x WMMA proj GEMM -> flash-attn (WMMA) -> WMMA out GEMM
// All intermediates are L2-resident (~130MB < 192MB L2).
// Round 2: 32 M-rows per wave everywhere so every B/k/v fragment load feeds
// two WMMAs (raises L2 arithmetic intensity ~2x; GEMM ~21 flop/B, attn ~32).
// ---------------------------------------------------------------------------

#define Bc 2
#define Sc 2048
#define Dc 1024
#define Hc 16
#define DHc 64

typedef __attribute__((ext_vector_type(16))) _Float16 v16h;
typedef __attribute__((ext_vector_type(8)))  _Float16 v8h;
typedef __attribute__((ext_vector_type(8)))  float    v8f;

union HalfFrag {
    v16h v;
    struct { v8h lo, hi; } h2;
};
union AccFrag {
    v8f   v;
    float e[8];
};

// Load a 16x32 f16 A-fragment (or B^T row-major == B fragment) per the CDNA5
// 16-bit A-matrix layout: lanes 0-15 hold row (row0+lane), K = {0..7,16..23};
// lanes 16-31 hold the same rows with K offset +8. Two b128 loads per lane.
__device__ __forceinline__ v16h load_frag16(const _Float16* __restrict__ base,
                                            int stride, int row0, int k0, int lane) {
    int r  = row0 + (lane & 15);
    int ko = k0 + ((lane & 16) ? 8 : 0);
    const _Float16* p = base + (long)r * stride + ko;
    HalfFrag f;
    f.h2.lo = *(const v8h*)(p);        // K = ko .. ko+7
    f.h2.hi = *(const v8h*)(p + 16);   // K = ko+16 .. ko+23
    return f.v;
}

__device__ __forceinline__ v8f wmma_f16(v16h a, v16h b, v8f c) {
    return __builtin_amdgcn_wmma_f32_16x16x32_f16(false, a, false, b,
                                                  (short)0, c, false, false);
}

// ---------------------------------------------------------------------------
// fp32 -> fp16 conversion, 8 elements/lane (b128-in / b128-out)
// ---------------------------------------------------------------------------
__global__ __launch_bounds__(256) void cvt_f32_f16(const float* __restrict__ in,
                                                   _Float16* __restrict__ out,
                                                   long n) {
    long i = ((long)blockIdx.x * blockDim.x + threadIdx.x) * 8;
    if (i + 7 < n) {
#pragma unroll
        for (int j = 0; j < 8; ++j) out[i + j] = (_Float16)in[i + j];
    }
}

// ---------------------------------------------------------------------------
// Y = X[M,K] @ W[N,K]^T.  One wave computes a 32x64 strip of Y
// (two 16-row tiles sharing every B fragment).
//   mode 0: f16 out, scatter to [B,H,S,DH]
//   mode 1: f16 out, scatter to [B,H,DH,S]   (v transposed for P@V B-frags)
//   mode 2: f32 out, plain [M,N]             (final projection to d_out)
// ---------------------------------------------------------------------------
__global__ __launch_bounds__(128) void gemm_xwT(const _Float16* __restrict__ X,
                                                const _Float16* __restrict__ W,
                                                _Float16* __restrict__ Yh,
                                                float* __restrict__ Yf,
                                                int M, int N, int K, int mode) {
    int lane = threadIdx.x & 31;
    int wave = threadIdx.x >> 5;
    int m0 = (blockIdx.x * 4 + wave) * 32;
    int n0 = blockIdx.y * 64;
    if (m0 >= M) return;

    AccFrag acc[2][4];
#pragma unroll
    for (int mt = 0; mt < 2; ++mt)
#pragma unroll
        for (int t = 0; t < 4; ++t)
#pragma unroll
            for (int r = 0; r < 8; ++r) acc[mt][t].e[r] = 0.0f;

    for (int k0 = 0; k0 < K; k0 += 32) {
        v16h a0 = load_frag16(X, K, m0,      k0, lane);
        v16h a1 = load_frag16(X, K, m0 + 16, k0, lane);
#pragma unroll
        for (int t = 0; t < 4; ++t) {
            v16h b = load_frag16(W, K, n0 + t * 16, k0, lane);
            acc[0][t].v = wmma_f16(a0, b, acc[0][t].v);
            acc[1][t].v = wmma_f16(a1, b, acc[1][t].v);
        }
    }

    int half16 = (lane & 16) ? 8 : 0;
    int nl = lane & 15;
#pragma unroll
    for (int mt = 0; mt < 2; ++mt) {
        if (mode == 2) {
#pragma unroll
            for (int t = 0; t < 4; ++t)
#pragma unroll
                for (int r = 0; r < 8; ++r) {
                    int m = m0 + mt * 16 + r + half16;
                    int n = n0 + t * 16 + nl;
                    Yf[(long)m * N + n] = acc[mt][t].e[r];
                }
        } else {
#pragma unroll
            for (int t = 0; t < 4; ++t)
#pragma unroll
                for (int r = 0; r < 8; ++r) {
                    int m = m0 + mt * 16 + r + half16;   // m = b*S + s
                    int n = n0 + t * 16 + nl;            // n = h*64 + dh
                    int b = m >> 11, s = m & (Sc - 1);
                    int h = n >> 6,  dh = n & 63;
                    long idx = (mode == 0)
                        ? (((long)(b * Hc + h) * Sc + s) * DHc + dh)
                        : (((long)(b * Hc + h) * DHc + dh) * Sc + s);
                    Yh[idx] = (_Float16)acc[mt][t].e[r];
                }
        }
    }
}

// ---------------------------------------------------------------------------
// Causal flash attention. One wave per 32-row q block (two 16-row tiles that
// share all k and v fragments); k processed in 64-wide blocks:
// 16 WMMA for scores, online softmax, P via LDS (f16), 16 WMMA for P@V.
// q,k: [B,H,S,DH] f16;  vT: [B,H,DH,S] f16;  AO out: [B*S, D] f16.
// ---------------------------------------------------------------------------
__global__ __launch_bounds__(128) void attn_fa(const _Float16* __restrict__ qb,
                                               const _Float16* __restrict__ kb,
                                               const _Float16* __restrict__ vTb,
                                               _Float16* __restrict__ AO) {
    __shared__ _Float16 lds[4][32 * 64];   // per-wave 32x64 P tile (4KB each)

    int lane = threadIdx.x & 31;
    int wave = threadIdx.x >> 5;
    int bh = blockIdx.y;                    // 0..B*H-1
    int b = bh >> 4, h = bh & 15;
    int q0 = (blockIdx.x * 4 + wave) * 32;

    const _Float16* qp = qb  + (long)(b * Hc + h) * Sc * DHc;
    const _Float16* kp = kb  + (long)(b * Hc + h) * Sc * DHc;
    const _Float16* vp = vTb + (long)(b * Hc + h) * DHc * Sc;
    _Float16* ldsw = &lds[wave][0];

    // q fragments: 2 m-tiles x 2 K-steps, kept in registers
    v16h qa[2][2];
#pragma unroll
    for (int mt = 0; mt < 2; ++mt) {
        qa[mt][0] = load_frag16(qp, DHc, q0 + mt * 16, 0,  lane);
        qa[mt][1] = load_frag16(qp, DHc, q0 + mt * 16, 32, lane);
    }

    AccFrag o[2][4];
#pragma unroll
    for (int mt = 0; mt < 2; ++mt)
#pragma unroll
        for (int t = 0; t < 4; ++t)
#pragma unroll
            for (int r = 0; r < 8; ++r) o[mt][t].e[r] = 0.0f;

    float rmax[2][8], rsum[2][8];
#pragma unroll
    for (int mt = 0; mt < 2; ++mt)
#pragma unroll
        for (int r = 0; r < 8; ++r) { rmax[mt][r] = -1e30f; rsum[mt][r] = 0.0f; }

    int half16 = (lane & 16) ? 8 : 0;
    int nl = lane & 15;
    int nblocks = (q0 + 32 + 63) >> 6;

    for (int jb = 0; jb < nblocks; ++jb) {
        int j0 = jb * 64;

        // ----- scores S = q @ k^T (32x64): k frags shared by both m tiles ---
        AccFrag sc[2][4];
#pragma unroll
        for (int t = 0; t < 4; ++t) {
#pragma unroll
            for (int mt = 0; mt < 2; ++mt)
#pragma unroll
                for (int r = 0; r < 8; ++r) sc[mt][t].e[r] = 0.0f;
            v16h b0 = load_frag16(kp, DHc, j0 + t * 16, 0,  lane);
            sc[0][t].v = wmma_f16(qa[0][0], b0, sc[0][t].v);
            sc[1][t].v = wmma_f16(qa[1][0], b0, sc[1][t].v);
            v16h b1 = load_frag16(kp, DHc, j0 + t * 16, 32, lane);
            sc[0][t].v = wmma_f16(qa[0][1], b1, sc[0][t].v);
            sc[1][t].v = wmma_f16(qa[1][1], b1, sc[1][t].v);
        }

        // ----- per m-tile: scale, mask, online softmax, P -> LDS -----------
#pragma unroll
        for (int mt = 0; mt < 2; ++mt) {
#pragma unroll
            for (int t = 0; t < 4; ++t)
#pragma unroll
                for (int r = 0; r < 8; ++r) {
                    int qq = q0 + mt * 16 + r + half16;
                    int kk = j0 + t * 16 + nl;
                    float v = sc[mt][t].e[r] * 0.125f;          // 1/sqrt(64)
                    sc[mt][t].e[r] = (kk <= qq) ? v : -1e30f;   // causal mask
                }

            float bm[8];
#pragma unroll
            for (int r = 0; r < 8; ++r) {
                float m = sc[mt][0].e[r];
#pragma unroll
                for (int t = 1; t < 4; ++t) m = fmaxf(m, sc[mt][t].e[r]);
                bm[r] = m;
            }
#pragma unroll
            for (int xm = 1; xm < 16; xm <<= 1)
#pragma unroll
                for (int r = 0; r < 8; ++r)
                    bm[r] = fmaxf(bm[r], __shfl_xor(bm[r], xm, 32));

            float alpha[8];
#pragma unroll
            for (int r = 0; r < 8; ++r) {
                float mnew = fmaxf(rmax[mt][r], bm[r]);
                alpha[r] = __expf(rmax[mt][r] - mnew);
                rmax[mt][r] = mnew;
            }

            float bs[8];
#pragma unroll
            for (int r = 0; r < 8; ++r) bs[r] = 0.0f;
#pragma unroll
            for (int t = 0; t < 4; ++t)
#pragma unroll
                for (int r = 0; r < 8; ++r) {
                    float p = __expf(sc[mt][t].e[r] - rmax[mt][r]);
                    sc[mt][t].e[r] = p;
                    bs[r] += p;
                }
#pragma unroll
            for (int xm = 1; xm < 16; xm <<= 1)
#pragma unroll
                for (int r = 0; r < 8; ++r)
                    bs[r] += __shfl_xor(bs[r], xm, 32);
#pragma unroll
            for (int r = 0; r < 8; ++r)
                rsum[mt][r] = rsum[mt][r] * alpha[r] + bs[r];

            // P (f16) to LDS (D-frag -> A-frag reshape)
#pragma unroll
            for (int t = 0; t < 4; ++t)
#pragma unroll
                for (int r = 0; r < 8; ++r)
                    ldsw[(mt * 16 + r + half16) * 64 + t * 16 + nl] =
                        (_Float16)sc[mt][t].e[r];

            // rescale O accumulators
#pragma unroll
            for (int t = 0; t < 4; ++t)
#pragma unroll
                for (int r = 0; r < 8; ++r) o[mt][t].e[r] *= alpha[r];
        }
        __asm__ volatile("s_wait_dscnt 0" ::: "memory");  // wave-local LDS RAW

        // ----- O += P @ V : v frags shared by both m tiles -----------------
        v16h pa[2][2];
#pragma unroll
        for (int mt = 0; mt < 2; ++mt) {
            pa[mt][0] = load_frag16(ldsw, 64, mt * 16, 0,  lane);
            pa[mt][1] = load_frag16(ldsw, 64, mt * 16, 32, lane);
        }
#pragma unroll
        for (int t = 0; t < 4; ++t) {
            v16h vb0 = load_frag16(vp, Sc, t * 16, j0,      lane);
            o[0][t].v = wmma_f16(pa[0][0], vb0, o[0][t].v);
            o[1][t].v = wmma_f16(pa[1][0], vb0, o[1][t].v);
            v16h vb1 = load_frag16(vp, Sc, t * 16, j0 + 32, lane);
            o[0][t].v = wmma_f16(pa[0][1], vb1, o[0][t].v);
            o[1][t].v = wmma_f16(pa[1][1], vb1, o[1][t].v);
        }
    }

    // ----- normalize, store AO[b*S + q, h*64 + dh] (f16) -----
#pragma unroll
    for (int mt = 0; mt < 2; ++mt)
#pragma unroll
        for (int t = 0; t < 4; ++t)
#pragma unroll
            for (int r = 0; r < 8; ++r) {
                int m = q0 + mt * 16 + r + half16;
                int n = h * 64 + t * 16 + nl;
                float val = o[mt][t].e[r] / rsum[mt][r];
                AO[(long)(b * Sc + m) * Dc + n] = (_Float16)val;
            }
}

// ---------------------------------------------------------------------------
extern "C" void kernel_launch(void* const* d_in, const int* in_sizes, int n_in,
                              void* d_out, int out_size, void* d_ws, size_t ws_size,
                              hipStream_t stream) {
    const float* Q  = (const float*)d_in[0];
    const float* K  = (const float*)d_in[1];
    const float* V  = (const float*)d_in[2];
    const float* wq = (const float*)d_in[3];
    const float* wk = (const float*)d_in[4];
    const float* wv = (const float*)d_in[5];
    const float* wo = (const float*)d_in[6];
    float* OUT = (float*)d_out;

    const long MS    = (long)Bc * Sc;   // 4096 rows
    const long X_ELE = MS * Dc;         // 4M elements
    const long W_ELE = (long)Dc * Dc;   // 1M elements

    // workspace carve-up (f16 staging, 64 MiB total)
    char* ws = (char*)d_ws;
    _Float16* Xq = (_Float16*)ws;  ws += X_ELE * 2;
    _Float16* Xk = (_Float16*)ws;  ws += X_ELE * 2;
    _Float16* Xv = (_Float16*)ws;  ws += X_ELE * 2;
    _Float16* Wq = (_Float16*)ws;  ws += W_ELE * 2;
    _Float16* Wk = (_Float16*)ws;  ws += W_ELE * 2;
    _Float16* Wv = (_Float16*)ws;  ws += W_ELE * 2;
    _Float16* Wo = (_Float16*)ws;  ws += W_ELE * 2;
    _Float16* qb = (_Float16*)ws;  ws += X_ELE * 2;   // [B,H,S,DH]
    _Float16* kb = (_Float16*)ws;  ws += X_ELE * 2;   // [B,H,S,DH]
    _Float16* vT = (_Float16*)ws;  ws += X_ELE * 2;   // [B,H,DH,S]
    _Float16* AO = (_Float16*)ws;  ws += X_ELE * 2;   // [B*S, D]

    // 1) fp32 -> fp16 staging
    cvt_f32_f16<<<dim3((unsigned)(X_ELE / (8 * 256))), 256, 0, stream>>>(Q,  Xq, X_ELE);
    cvt_f32_f16<<<dim3((unsigned)(X_ELE / (8 * 256))), 256, 0, stream>>>(K,  Xk, X_ELE);
    cvt_f32_f16<<<dim3((unsigned)(X_ELE / (8 * 256))), 256, 0, stream>>>(V,  Xv, X_ELE);
    cvt_f32_f16<<<dim3((unsigned)(W_ELE / (8 * 256))), 256, 0, stream>>>(wq, Wq, W_ELE);
    cvt_f32_f16<<<dim3((unsigned)(W_ELE / (8 * 256))), 256, 0, stream>>>(wk, Wk, W_ELE);
    cvt_f32_f16<<<dim3((unsigned)(W_ELE / (8 * 256))), 256, 0, stream>>>(wv, Wv, W_ELE);
    cvt_f32_f16<<<dim3((unsigned)(W_ELE / (8 * 256))), 256, 0, stream>>>(wo, Wo, W_ELE);

    // 2) q/k/v projections (WMMA GEMM, 32x64 per wave); v stored transposed
    dim3 gblk(128);
    dim3 ggrd((unsigned)(MS / 128), (unsigned)(Dc / 64));
    gemm_xwT<<<ggrd, gblk, 0, stream>>>(Xq, Wq, qb, nullptr, (int)MS, Dc, Dc, 0);
    gemm_xwT<<<ggrd, gblk, 0, stream>>>(Xk, Wk, kb, nullptr, (int)MS, Dc, Dc, 0);
    gemm_xwT<<<ggrd, gblk, 0, stream>>>(Xv, Wv, vT, nullptr, (int)MS, Dc, Dc, 1);

    // 3) causal flash attention (WMMA, 32 q-rows per wave)
    attn_fa<<<dim3(Sc / 128, Bc * Hc), 128, 0, stream>>>(qb, kb, vT, AO);

    // 4) output projection to fp32 d_out
    gemm_xwT<<<ggrd, gblk, 0, stream>>>(AO, Wo, nullptr, OUT, (int)MS, Dc, Dc, 2);
}
